// RNADegformer_30666066493924
// MI455X (gfx1250) — matrix operand order repaired and checked by reading, not compiled, not run on hardware
//
#include <hip/hip_runtime.h>
#include <hip/hip_bf16.h>

// ---------------------------------------------------------------------------
// RNADegformer forward for MI455X (gfx1250).
// - All large matmuls via v_wmma_f32_16x16x32_bf16 (bf16 A/B, f32 accum).
// - LDS tiles stored in fragment order: every WMMA operand = 2x ds_load_b128.
// - Weight B-tiles staged to LDS by the Tensor Data Mover (tensor_load_to_lds
//   + s_wait_tensorcnt); software-pipelined double buffering overlaps TDM and
//   global b128 loads with the WMMA stream.
// - fp32->bf16 RNE conversion packs each pair with one v_perm_b32.
// ---------------------------------------------------------------------------

typedef __attribute__((ext_vector_type(16))) __bf16          v16bf;
typedef __attribute__((ext_vector_type(2)))  __bf16          v2bf;
typedef __attribute__((ext_vector_type(8)))  float           v8f;
typedef __attribute__((ext_vector_type(4)))  float           f32x4;
typedef __attribute__((ext_vector_type(2)))  unsigned int    u32x2;
typedef __attribute__((ext_vector_type(4)))  unsigned int    u32x4;
typedef __attribute__((ext_vector_type(8)))  int             i32x8;
typedef __attribute__((ext_vector_type(4)))  int             i32x4;

#define B_   8
#define S_   512
#define D_   768
#define H_   12
#define DK_  64
#define F_   3072
#define L_   4
#define NC_  5
#define E_   480
#define TOK_ (B_ * S_)   // 4096 rows

#if defined(__AMDGCN__) && __has_builtin(__builtin_amdgcn_tensor_load_to_lds) && \
    __has_builtin(__builtin_amdgcn_s_wait_tensorcnt)
#define HAVE_TDM 1
#else
#define HAVE_TDM 0
#endif

// ---- fp32 -> bf16 (round to nearest even) ---------------------------------
__device__ __forceinline__ unsigned short f2bf(float f) {
    union { float f; unsigned u; } v; v.f = f;
    unsigned u = v.u;
    unsigned r = u + 0x7FFFu + ((u >> 16) & 1u);
    return (unsigned short)(r >> 16);
}

// packed pair conversion: prefer hw packed cvt, else RNE + single v_perm_b32
__device__ __forceinline__ unsigned pack2bf(float a, float b) {
#if defined(__AMDGCN__) && __has_builtin(__builtin_amdgcn_cvt_pk_bf16_f32)
    v2bf r = __builtin_amdgcn_cvt_pk_bf16_f32(a, b);
    return __builtin_bit_cast(unsigned, r);
#elif defined(__AMDGCN__) && __has_builtin(__builtin_amdgcn_perm)
    unsigned ua = __builtin_bit_cast(unsigned, a);
    unsigned ub = __builtin_bit_cast(unsigned, b);
    ua = ua + 0x7FFFu + ((ua >> 16) & 1u);   // RNE, result in high half
    ub = ub + 0x7FFFu + ((ub >> 16) & 1u);
    // result = { ub[31:16], ua[31:16] } in one V_PERM_B32
    return __builtin_amdgcn_perm(ub, ua, 0x07060302u);
#else
    return (unsigned)f2bf(a) | ((unsigned)f2bf(b) << 16);
#endif
}

// fp32 [K,N] -> bf16 transposed [N,K]  (one-time weight prep)
__global__ void cvt_bf16_t_kernel(const float* __restrict__ src,
                                  unsigned short* __restrict__ dst,
                                  int K, int N) {
    int i = blockIdx.x * blockDim.x + threadIdx.x;
    if (i >= K * N) return;
    int k = i / N, n = i - k * N;
    dst[(size_t)n * K + k] = f2bf(src[i]);
}

// ---- fragment helpers -----------------------------------------------------
union AFrag { v16bf v; u32x4 q[2]; };

// A tile in LDS: 64 rows x 40 ushort (row-major, K contiguous, 80B stride).
__device__ __forceinline__ void load_a_frag(AFrag& af, const unsigned short* As,
                                            int ra, int half) {
    const u32x4* arow = (const u32x4*)(As + (size_t)ra * 40);
    af.q[0] = arow[half];
    af.q[1] = arow[2 + half];
}
// B tile in LDS, TRANSPOSED: 64 cols x 40 ushort (K contiguous per column).
__device__ __forceinline__ void load_b_frag(AFrag& bf, const unsigned short* Bt,
                                            int col, int half) {
    const u32x4* brow = (const u32x4*)(Bt + (size_t)col * 40);
    bf.q[0] = brow[half];
    bf.q[1] = brow[2 + half];
}

#if HAVE_TDM
// TDM: 64-row x 32-elem bf16 tile from WT[N][K] (row stride K elems) into LDS
// with +16B pad every 64B -> 80B LDS row stride.
__device__ __forceinline__ void tdm_load_b_tile(const unsigned short* gsrc,
                                                unsigned lds_addr,
                                                int K, int N) {
    unsigned long long ga = (unsigned long long)(size_t)gsrc;
    u32x4 g0;
    g0[0] = 1u;                                               // count=1 (valid)
    g0[1] = lds_addr;                                         // LDS byte address
    g0[2] = (unsigned)(ga & 0xFFFFFFFFu);                     // global_addr[31:0]
    g0[3] = (unsigned)((ga >> 32) & 0x1FFFFFFu) | (2u << 30); // [56:32] | type=2
    i32x8 g1;
    g1[0] = (1 << 16) | (1 << 20) | (3 << 22) | (3 << 25);    // 2B elems, pad 16DW->+4DW
    g1[1] = (int)(((unsigned)K & 0xFFFFu) << 16);             // tensor_dim0[15:0]
    g1[2] = (int)((((unsigned)K >> 16) & 0xFFFFu) |
                  (((unsigned)N & 0xFFFFu) << 16));           // dim0[31:16] | dim1[15:0]
    g1[3] = (int)((((unsigned)N >> 16) & 0xFFFFu) |
                  (32u << 16));                               // dim1[31:16] | tile_dim0=32
    g1[4] = 64;                                               // tile_dim1=64
    g1[5] = K;                                                // tensor_dim0_stride
    g1[6] = 0;
    g1[7] = 0;
    i32x4 gz = {0, 0, 0, 0};
#if defined(__clang_major__) && (__clang_major__ >= 23)
    i32x8 z8 = {0, 0, 0, 0, 0, 0, 0, 0};
    __builtin_amdgcn_tensor_load_to_lds(g0, g1, gz, gz, z8, 0);
#else
    __builtin_amdgcn_tensor_load_to_lds(g0, g1, gz, gz, 0);
#endif
}
#endif

// ---------------------------------------------------------------------------
// GEMM: C[M,N] = A[M,K](f32) x WT[N,K](bf16, pre-transposed) + bias (+relu)
// Block = 128 threads (4 waves), tile 64x64, K-step 32, double-buffered.
// ---------------------------------------------------------------------------
__launch_bounds__(128)
__global__ void gemm_bf16_kernel(const float* __restrict__ A, int lda,
                                 const unsigned short* __restrict__ WT,
                                 const float* __restrict__ bias,
                                 float* __restrict__ C, int N,
                                 int K, int relu) {
    __shared__ unsigned short As[2][64][40] __attribute__((aligned(16)));
    __shared__ unsigned short Bt[2][64][40] __attribute__((aligned(16)));
    const int tid  = threadIdx.x;
    const int lane = tid & 31;
    const int wave = tid >> 5;
    const int half = lane >> 4;
    const int nn   = lane & 15;
    const int m0   = blockIdx.y * 64;
    const int n0   = blockIdx.x * 64;

    v8f acc[4];
    #pragma unroll
    for (int t = 0; t < 4; ++t)
        #pragma unroll
        for (int j = 0; j < 8; ++j) acc[t][j] = 0.0f;

    f32x4 fa[4];
    auto a_load = [&](int k0) {
        #pragma unroll
        for (int e = 0; e < 4; ++e) {
            int idx = tid + 128 * e;
            int r = idx >> 3, c4 = idx & 7;
            fa[e] = *(const f32x4*)(A + (size_t)(m0 + r) * lda + k0 + 4 * c4);
        }
    };
    auto a_store = [&](int buf) {
        #pragma unroll
        for (int e = 0; e < 4; ++e) {
            int idx = tid + 128 * e;
            int r = idx >> 3, c4 = idx & 7;
            u32x2 p;
            p[0] = pack2bf(fa[e][0], fa[e][1]);
            p[1] = pack2bf(fa[e][2], fa[e][3]);
            *(u32x2*)&As[buf][r][4 * c4] = p;
        }
    };
    auto b_stage = [&](int buf, int k0) {
#if HAVE_TDM
        if (wave == 0)
            tdm_load_b_tile(WT + (size_t)n0 * K + k0,
                            (unsigned)(size_t)&Bt[buf][0][0], K, N);
#else
        #pragma unroll
        for (int e = 0; e < 2; ++e) {
            int idx = tid + 128 * e;
            int c = idx >> 2, r8 = idx & 3;
            *(u32x4*)&Bt[buf][c][8 * r8] =
                *(const u32x4*)(WT + (size_t)(n0 + c) * K + k0 + 8 * r8);
        }
#endif
    };

    // prologue: stage tile 0
    b_stage(0, 0);
    a_load(0);
    a_store(0);
#if HAVE_TDM
    if (wave == 0) __builtin_amdgcn_s_wait_tensorcnt(0);
#endif
    __syncthreads();

    const int nsteps = K / 32;
    for (int s = 0; s < nsteps; ++s) {
        const int cur = s & 1, nxt = cur ^ 1;
        const bool more = (s + 1) < nsteps;
        if (more) {                      // kick off next tile's memory early
            b_stage(nxt, (s + 1) * 32);
            a_load((s + 1) * 32);
        }
        AFrag af;                        // compute current tile
        load_a_frag(af, &As[cur][0][0], 16 * wave + nn, half);
        #pragma unroll
        for (int t = 0; t < 4; ++t) {
            AFrag bf;
            load_b_frag(bf, &Bt[cur][0][0], 16 * t + nn, half);
            acc[t] = __builtin_amdgcn_wmma_f32_16x16x32_bf16(
                false, af.v, false, bf.v, (short)0, acc[t], false, false);
        }
        if (more) {
            a_store(nxt);
#if HAVE_TDM
            if (wave == 0) __builtin_amdgcn_s_wait_tensorcnt(0);
#endif
        }
        __syncthreads();
    }

    #pragma unroll
    for (int t = 0; t < 4; ++t)
        #pragma unroll
        for (int j = 0; j < 8; ++j) {
            int row = m0 + 16 * wave + 8 * half + j;
            int col = n0 + 16 * t + nn;
            float v = acc[t][j];
            if (bias) v += bias[col];
            if (relu) v = v > 0.0f ? v : 0.0f;
            C[(size_t)row * N + col] = v;
        }
}

// ---------------------------------------------------------------------------
// Attention scores: attn[b,h,i,j] = (Q.K^T)/8 + gamma*(bpp.mdw + mdb), masked
// Per (b,h): M=N=512, K=DK=64 (2 K-steps). Tile 64x64, grid (8,8,B*H).
// ---------------------------------------------------------------------------
__launch_bounds__(128)
__global__ void attn_scores_kernel(const float* __restrict__ q,
                                   const float* __restrict__ kmat,
                                   const float* __restrict__ bpp,
                                   const float* __restrict__ mdwL,   // [H,4]
                                   const float* __restrict__ mdbL,   // [H]
                                   const float* __restrict__ gammaL, // [1]
                                   const int*   __restrict__ src_mask, // [B,L,S]
                                   float* __restrict__ attn, int layer) {
    __shared__ unsigned short As[64][40] __attribute__((aligned(16)));
    __shared__ unsigned short Bt[64][40] __attribute__((aligned(16)));
    const int tid = threadIdx.x, lane = tid & 31, wave = tid >> 5;
    const int half = lane >> 4, nn = lane & 15;
    const int z = blockIdx.z, b = z / H_, h = z % H_;
    const int i0 = blockIdx.y * 64, j0 = blockIdx.x * 64;
    const float gamma = gammaL[0];
    const float w0 = mdwL[h * 4 + 0], w1 = mdwL[h * 4 + 1];
    const float w2 = mdwL[h * 4 + 2], w3 = mdwL[h * 4 + 3];
    const float mb = mdbL[h];

    v8f acc[4];
    #pragma unroll
    for (int t = 0; t < 4; ++t)
        #pragma unroll
        for (int j = 0; j < 8; ++j) acc[t][j] = 0.0f;

    for (int k0 = 0; k0 < DK_; k0 += 32) {
        #pragma unroll
        for (int e = 0; e < 4; ++e) {               // Q tile 64x32
            int idx = tid + 128 * e;
            int r = idx >> 3, c4 = idx & 7;
            f32x4 f = *(const f32x4*)(q + ((size_t)(b * S_ + i0 + r)) * D_ +
                                      h * DK_ + k0 + 4 * c4);
            u32x2 p; p[0] = pack2bf(f[0], f[1]); p[1] = pack2bf(f[2], f[3]);
            *(u32x2*)&As[r][4 * c4] = p;
        }
        #pragma unroll
        for (int e = 0; e < 4; ++e) {               // K tile, K-contig per col
            int idx = tid + 128 * e;
            int n = idx >> 3, c4 = idx & 7;
            f32x4 f = *(const f32x4*)(kmat + ((size_t)(b * S_ + j0 + n)) * D_ +
                                      h * DK_ + k0 + 4 * c4);
            u32x2 p; p[0] = pack2bf(f[0], f[1]); p[1] = pack2bf(f[2], f[3]);
            *(u32x2*)&Bt[n][4 * c4] = p;
        }
        __syncthreads();

        AFrag af;
        load_a_frag(af, &As[0][0], 16 * wave + nn, half);
        #pragma unroll
        for (int t = 0; t < 4; ++t) {
            AFrag bf;
            load_b_frag(bf, &Bt[0][0], 16 * t + nn, half);
            acc[t] = __builtin_amdgcn_wmma_f32_16x16x32_bf16(
                false, af.v, false, bf.v, (short)0, acc[t], false, false);
        }
        __syncthreads();
    }

    #pragma unroll
    for (int t = 0; t < 4; ++t)
        #pragma unroll
        for (int j = 0; j < 8; ++j) {
            int i  = i0 + 16 * wave + 8 * half + j;
            int jj = j0 + 16 * t + nn;
            float v = acc[t][j] * 0.125f;           // / sqrt(DK)
            size_t pb = (((size_t)b * 4) * S_ + i) * S_ + jj;
            float m = mb + w0 * bpp[pb]
                         + w1 * bpp[pb + (size_t)S_ * S_]
                         + w2 * bpp[pb + (size_t)2 * S_ * S_]
                         + w3 * bpp[pb + (size_t)3 * S_ * S_];
            v += gamma * m;
            int mi = src_mask[((size_t)b * L_ + layer) * S_ + i];
            int mj = src_mask[((size_t)b * L_ + layer) * S_ + jj];
            if ((mi & mj) == 0) v = -1e-9f;
            attn[((size_t)z * S_ + i) * S_ + jj] = v;
        }
}

// ---- softmax over rows of length 512 --------------------------------------
__launch_bounds__(256)
__global__ void softmax512_kernel(float* __restrict__ attn) {
    __shared__ float red[256];
    float* p = attn + (size_t)blockIdx.x * S_;
    int t = threadIdx.x;
    float a = p[t], b = p[t + 256];
    red[t] = fmaxf(a, b); __syncthreads();
    for (int s = 128; s > 0; s >>= 1) { if (t < s) red[t] = fmaxf(red[t], red[t + s]); __syncthreads(); }
    float mx = red[0]; __syncthreads();
    float ea = __expf(a - mx), eb = __expf(b - mx);
    red[t] = ea + eb; __syncthreads();
    for (int s = 128; s > 0; s >>= 1) { if (t < s) red[t] += red[t + s]; __syncthreads(); }
    float inv = 1.0f / red[0];
    p[t] = ea * inv; p[t + 256] = eb * inv;
}

// ---------------------------------------------------------------------------
// O = A x V per (b,h): M=512, N=64, K=512 (16 K-steps, double-buffered).
// ---------------------------------------------------------------------------
__launch_bounds__(128)
__global__ void attn_av_kernel(const float* __restrict__ attn,
                               const float* __restrict__ vmat,
                               float* __restrict__ o) {
    __shared__ unsigned short As[2][64][40] __attribute__((aligned(16)));
    __shared__ unsigned short Bt[2][64][40] __attribute__((aligned(16)));
    const int tid = threadIdx.x, lane = tid & 31, wave = tid >> 5;
    const int half = lane >> 4, nn = lane & 15;
    const int z = blockIdx.z, b = z / H_, h = z % H_;
    const int i0 = blockIdx.y * 64;

    v8f acc[4];
    #pragma unroll
    for (int t = 0; t < 4; ++t)
        #pragma unroll
        for (int j = 0; j < 8; ++j) acc[t][j] = 0.0f;

    f32x4 fa[4], fv[4];
    auto ld = [&](int k0) {
        #pragma unroll
        for (int e = 0; e < 4; ++e) {               // probs tile 64x32
            int idx = tid + 128 * e;
            int r = idx >> 3, c4 = idx & 7;
            fa[e] = *(const f32x4*)(attn + ((size_t)z * S_ + i0 + r) * S_ +
                                    k0 + 4 * c4);
        }
        #pragma unroll
        for (int e = 0; e < 4; ++e) {               // V tile (to transpose)
            int idx = tid + 128 * e;
            int nq = idx & 15, kk = idx >> 4;
            fv[e] = *(const f32x4*)(vmat + ((size_t)(b * S_ + k0 + kk)) * D_ +
                                    h * DK_ + 4 * nq);
        }
    };
    auto st = [&](int buf) {
        #pragma unroll
        for (int e = 0; e < 4; ++e) {
            int idx = tid + 128 * e;
            int r = idx >> 3, c4 = idx & 7;
            u32x2 p; p[0] = pack2bf(fa[e][0], fa[e][1]);
                     p[1] = pack2bf(fa[e][2], fa[e][3]);
            *(u32x2*)&As[buf][r][4 * c4] = p;
        }
        #pragma unroll
        for (int e = 0; e < 4; ++e) {
            int idx = tid + 128 * e;
            int nq = idx & 15, kk = idx >> 4;
            Bt[buf][4 * nq + 0][kk] = f2bf(fv[e][0]);
            Bt[buf][4 * nq + 1][kk] = f2bf(fv[e][1]);
            Bt[buf][4 * nq + 2][kk] = f2bf(fv[e][2]);
            Bt[buf][4 * nq + 3][kk] = f2bf(fv[e][3]);
        }
    };

    ld(0); st(0);
    __syncthreads();

    const int nsteps = S_ / 32;
    for (int s = 0; s < nsteps; ++s) {
        const int cur = s & 1, nxt = cur ^ 1;
        const bool more = (s + 1) < nsteps;
        if (more) ld((s + 1) * 32);
        AFrag af;
        load_a_frag(af, &As[cur][0][0], 16 * wave + nn, half);
        #pragma unroll
        for (int t = 0; t < 4; ++t) {
            AFrag bf;
            load_b_frag(bf, &Bt[cur][0][0], 16 * t + nn, half);
            acc[t] = __builtin_amdgcn_wmma_f32_16x16x32_bf16(
                false, af.v, false, bf.v, (short)0, acc[t], false, false);
        }
        if (more) st(nxt);
        __syncthreads();
    }

    #pragma unroll
    for (int t = 0; t < 4; ++t)
        #pragma unroll
        for (int j = 0; j < 8; ++j) {
            int row = i0 + 16 * wave + 8 * half + j;
            int col = 16 * t + nn;                  // 0..63 = dk
            o[((size_t)(b * S_ + row)) * D_ + h * DK_ + col] = acc[t][j];
        }
}

// ---- LayerNorm over D=768 with optional fused residual --------------------
__launch_bounds__(256)
__global__ void ln768_kernel(const float* __restrict__ a,
                             const float* __restrict__ r,
                             float* __restrict__ out,
                             const float* __restrict__ s,
                             const float* __restrict__ bi) {
    __shared__ float red[256];
    const size_t row = blockIdx.x;
    const int t = threadIdx.x;
    float v[3];
    #pragma unroll
    for (int e = 0; e < 3; ++e) {
        int c = t + 256 * e;
        v[e] = a[row * D_ + c] + (r ? r[row * D_ + c] : 0.0f);
    }
    red[t] = v[0] + v[1] + v[2]; __syncthreads();
    for (int st = 128; st > 0; st >>= 1) { if (t < st) red[t] += red[t + st]; __syncthreads(); }
    const float mean = red[0] * (1.0f / D_); __syncthreads();
    float sq = 0.0f;
    #pragma unroll
    for (int e = 0; e < 3; ++e) { float d = v[e] - mean; sq += d * d; }
    red[t] = sq; __syncthreads();
    for (int st = 128; st > 0; st >>= 1) { if (t < st) red[t] += red[t + st]; __syncthreads(); }
    const float inv = rsqrtf(red[0] * (1.0f / D_) + 1e-5f);
    #pragma unroll
    for (int e = 0; e < 3; ++e) {
        int c = t + 256 * e;
        out[row * D_ + c] = (v[e] - mean) * inv * s[c] + bi[c];
    }
}

// ---- decoder: [TOK,768] x [768,5] + b (too narrow for WMMA) ---------------
__global__ void decoder_kernel(const float* __restrict__ x,
                               const float* __restrict__ w,
                               const float* __restrict__ b,
                               float* __restrict__ out) {
    int i = blockIdx.x * blockDim.x + threadIdx.x;
    if (i >= TOK_ * NC_) return;
    int row = i / NC_, c = i % NC_;
    float acc = b[c];
    const f32x4* xr = (const f32x4*)(x + (size_t)row * D_);
    for (int d = 0; d < D_ / 4; ++d) {
        f32x4 f = xr[d];
        acc += f[0] * w[(4 * d + 0) * NC_ + c] + f[1] * w[(4 * d + 1) * NC_ + c] +
               f[2] * w[(4 * d + 2) * NC_ + c] + f[3] * w[(4 * d + 3) * NC_ + c];
    }
    out[i] = acc;
}

// ---------------------------------------------------------------------------
extern "C" void kernel_launch(void* const* d_in, const int* in_sizes, int n_in,
                              void* d_out, int out_size, void* d_ws, size_t ws_size,
                              hipStream_t stream) {
    (void)in_sizes; (void)n_in; (void)out_size; (void)ws_size;
    const float* rep      = (const float*)d_in[0];
    const float* bpp      = (const float*)d_in[1];
    const int*   src_mask = (const int*)  d_in[2];
    const float* proj_w   = (const float*)d_in[3];
    const float* proj_b   = (const float*)d_in[4];
    const float* pln_s    = (const float*)d_in[5];
    const float* pln_b    = (const float*)d_in[6];
    const float* Win      = (const float*)d_in[7];
    const float* bin_     = (const float*)d_in[8];
    const float* ln3s     = (const float*)d_in[9];
    const float* ln3b     = (const float*)d_in[10];
    const float* mdw      = (const float*)d_in[11];
    const float* mdb      = (const float*)d_in[12];
    const float* wq       = (const float*)d_in[13];
    const float* wk       = (const float*)d_in[14];
    const float* wv       = (const float*)d_in[15];
    const float* gamma    = (const float*)d_in[16];
    const float* ln1s     = (const float*)d_in[17];
    const float* ln1b     = (const float*)d_in[18];
    const float* W1       = (const float*)d_in[19];
    const float* b1       = (const float*)d_in[20];
    const float* W2       = (const float*)d_in[21];
    const float* b2       = (const float*)d_in[22];
    const float* ln2s     = (const float*)d_in[23];
    const float* ln2b     = (const float*)d_in[24];
    const float* Wout     = (const float*)d_in[25];
    const float* bout     = (const float*)d_in[26];
    const float* ln4s     = (const float*)d_in[27];
    const float* ln4b     = (const float*)d_in[28];
    const float* dec_w    = (const float*)d_in[29];
    const float* dec_b    = (const float*)d_in[30];

    // ---- workspace carve-out ----
    char* ws = (char*)d_ws;
    size_t off = 0;
    auto carve = [&](size_t bytes) -> void* {
        void* p = ws + off;
        off = (off + bytes + 255) & ~(size_t)255;
        return p;
    };
    unsigned short* proj_wb = (unsigned short*)carve((size_t)E_ * D_ * 2);
    unsigned short* Winb    = (unsigned short*)carve((size_t)L_ * D_ * D_ * 2);
    unsigned short* wqb     = (unsigned short*)carve((size_t)L_ * D_ * D_ * 2);
    unsigned short* wkb     = (unsigned short*)carve((size_t)L_ * D_ * D_ * 2);
    unsigned short* wvb     = (unsigned short*)carve((size_t)L_ * D_ * D_ * 2);
    unsigned short* W1b     = (unsigned short*)carve((size_t)L_ * D_ * F_ * 2);
    unsigned short* W2b     = (unsigned short*)carve((size_t)L_ * F_ * D_ * 2);
    unsigned short* Woutb   = (unsigned short*)carve((size_t)L_ * D_ * D_ * 2);
    float* xbuf  = (float*)carve((size_t)TOK_ * D_ * 4);
    float* hbuf  = (float*)carve((size_t)TOK_ * D_ * 4);
    float* tmp   = (float*)carve((size_t)TOK_ * D_ * 4);
    float* qbuf  = (float*)carve((size_t)TOK_ * D_ * 4);
    float* kbuf  = (float*)carve((size_t)TOK_ * D_ * 4);
    float* vbuf  = (float*)carve((size_t)TOK_ * D_ * 4);
    float* obuf  = (float*)carve((size_t)TOK_ * D_ * 4);
    float* ffbuf = (float*)carve((size_t)TOK_ * F_ * 4);
    float* attnb = (float*)carve((size_t)B_ * H_ * S_ * S_ * 4);

    // one-time bf16 conversion, TRANSPOSED: WT[N][K]
    auto cvtT = [&](const float* s, unsigned short* d, int K, int N) {
        int n = K * N;
        cvt_bf16_t_kernel<<<(n + 255) / 256, 256, 0, stream>>>(s, d, K, N);
    };
    cvtT(proj_w, proj_wb, E_, D_);
    for (int i = 0; i < L_; ++i) {
        cvtT(Win  + (size_t)i * D_ * D_, Winb  + (size_t)i * D_ * D_, D_, D_);
        cvtT(wq   + (size_t)i * D_ * D_, wqb   + (size_t)i * D_ * D_, D_, D_);
        cvtT(wk   + (size_t)i * D_ * D_, wkb   + (size_t)i * D_ * D_, D_, D_);
        cvtT(wv   + (size_t)i * D_ * D_, wvb   + (size_t)i * D_ * D_, D_, D_);
        cvtT(W1   + (size_t)i * D_ * F_, W1b   + (size_t)i * D_ * F_, D_, F_);
        cvtT(W2   + (size_t)i * F_ * D_, W2b   + (size_t)i * F_ * D_, F_, D_);
        cvtT(Wout + (size_t)i * D_ * D_, Woutb + (size_t)i * D_ * D_, D_, D_);
    }

    auto gemm = [&](const float* A, int lda, const unsigned short* WT,
                    const float* bias, float* C, int M, int N, int K, int relu) {
        dim3 g(N / 64, M / 64, 1);
        gemm_bf16_kernel<<<g, 128, 0, stream>>>(A, lda, WT, bias, C, N, K, relu);
    };
    auto ln = [&](const float* a, const float* r, float* o, const float* s, const float* b) {
        ln768_kernel<<<TOK_, 256, 0, stream>>>(a, r, o, s, b);
    };

    // projection + LN
    gemm(rep, E_, proj_wb, proj_b, tmp, TOK_, D_, E_, 0);
    ln(tmp, nullptr, xbuf, pln_s, pln_b);

    for (int i = 0; i < L_; ++i) {
        // h = LN(x @ Win + bin)
        gemm(xbuf, D_, Winb + (size_t)i * D_ * D_, bin_ + i * D_, tmp, TOK_, D_, D_, 0);
        ln(tmp, nullptr, hbuf, ln3s + i * D_, ln3b + i * D_);
        // q,k,v
        gemm(hbuf, D_, wqb + (size_t)i * D_ * D_, nullptr, qbuf, TOK_, D_, D_, 0);
        gemm(hbuf, D_, wkb + (size_t)i * D_ * D_, nullptr, kbuf, TOK_, D_, D_, 0);
        gemm(hbuf, D_, wvb + (size_t)i * D_ * D_, nullptr, vbuf, TOK_, D_, D_, 0);
        // attention
        attn_scores_kernel<<<dim3(S_ / 64, S_ / 64, B_ * H_), 128, 0, stream>>>(
            qbuf, kbuf, bpp, mdw + i * H_ * 4, mdb + i * H_, gamma + i, src_mask, attnb, i);
        softmax512_kernel<<<B_ * H_ * S_, 256, 0, stream>>>(attnb);
        attn_av_kernel<<<dim3(1, S_ / 64, B_ * H_), 128, 0, stream>>>(attnb, vbuf, obuf);
        // h = LN(h + o)
        ln(hbuf, obuf, hbuf, ln1s + i * D_, ln1b + i * D_);
        // FFN
        gemm(hbuf, D_, W1b + (size_t)i * D_ * F_, b1 + i * F_, ffbuf, TOK_, F_, D_, 1);
        gemm(ffbuf, F_, W2b + (size_t)i * F_ * D_, b2 + i * D_, tmp, TOK_, D_, F_, 0);
        ln(hbuf, tmp, hbuf, ln2s + i * D_, ln2b + i * D_);
        // x = LN(res + h @ Wout + bout)
        gemm(hbuf, D_, Woutb + (size_t)i * D_ * D_, bout + i * D_, tmp, TOK_, D_, D_, 0);
        ln(xbuf, tmp, xbuf, ln4s + i * D_, ln4b + i * D_);
    }

    decoder_kernel<<<(TOK_ * NC_ + 255) / 256, 256, 0, stream>>>(xbuf, dec_w, dec_b, (float*)d_out);
}
